// MNISTIRNNPyTorch_71889162600449
// MI455X (gfx1250) — matrix operand order repaired and checked by reading, not compiled, not run on hardware
//
#include <hip/hip_runtime.h>

typedef _Float16 v16h __attribute__((ext_vector_type(16)));
typedef _Float16 v8h  __attribute__((ext_vector_type(8)));
typedef float    v8f  __attribute__((ext_vector_type(8)));

#define TSTEPS  784
#define HDIM    256
#define BM      16      // batch rows per block
#define HPAD    264     // padded LDS row stride (halves) -> 528B, breaks bank conflicts
#define NBLOCKS 32      // 32 * 16 = 512 = B
#define NTHREADS 256    // 8 wave32 waves

__global__ __launch_bounds__(NTHREADS, 1)
void irnn_persistent_kernel(const float* __restrict__ x,      // [512,784]
                            const float* __restrict__ W_ih,   // [256,1]
                            const float* __restrict__ b_ih,   // [256]
                            const float* __restrict__ W_hh,   // [256,256]
                            const float* __restrict__ b_hh,   // [256]
                            const float* __restrict__ W_out,  // [10,256]
                            const float* __restrict__ b_out,  // [10]
                            float* __restrict__ out)          // [512,10]
{
    __shared__ _Float16 hbuf[2][BM][HPAD];   // double-buffered hidden state (f16)
    __shared__ float    hf[BM][257];         // final hidden state (f32), padded
    __shared__ float    logitsS[BM][10];

    const int tid    = threadIdx.x;
    const int lane   = tid & 31;
    const int wave   = tid >> 5;        // 0..7
    const int laneLo = lane & 15;       // col (B/C/D) or row (A) index
    const int grp    = lane >> 4;       // half-wave group
    const int bBase  = blockIdx.x * BM; // this block's batch rows

    // ---- zero h0 ----
    {
        _Float16* z = &hbuf[0][0][0];
        for (int i = tid; i < 2 * BM * HPAD; i += NTHREADS) z[i] = (_Float16)0.0f;
    }

    // ---- preload this wave's W_hh slice into registers (f32 -> f16), plus
    //      per-lane input-projection weight and fused bias (loop invariants) ----
    const int j0 = wave * 32;           // wave handles H columns [j0, j0+32)
    v16h  Breg[2][8];
    float wj[2], bj[2];
#pragma unroll
    for (int tile = 0; tile < 2; ++tile) {
        const int j = j0 + tile * 16 + laneLo;
        wj[tile] = W_ih[j];                      // W_ih is [H,1] flat
        bj[tile] = b_ih[j] + b_hh[j];
        // B operand, this code's own k order (same as A): lane holds col n = lane&15, contiguous K chunk
        // starting at grp*16 within each 32-wide K block.  Row j of W_hh,
        // loaded row-major, IS column j of B = W_hh^T.
        const float* wrow = W_hh + j * HDIM + grp * 16;
#pragma unroll
        for (int kb = 0; kb < 8; ++kb) {
            v16h bv;
#pragma unroll
            for (int e = 0; e < 16; ++e)
                bv[e] = (_Float16)wrow[kb * 32 + e];
            Breg[tile][kb] = bv;
        }
    }
    __syncthreads();

    const int rbase = grp * 8;          // C/D rows this lane holds

    // ---- recurrence: 784 sequential steps, all state in LDS + VGPRs ----
#pragma unroll 1
    for (int t = 0; t < TSTEPS; ++t) {
        const int cur = t & 1;
        const int nxt = cur ^ 1;

        v8f acc0 = {};
        v8f acc1 = {};
#pragma unroll
        for (int kb = 0; kb < 8; ++kb) {
            // A operand, this code's own k order (same as B): lane holds row m = lane&15, one contiguous
            // 16-half chunk at K = kb*32 + grp*16.
            const v8h alo = *(const v8h*)&hbuf[cur][laneLo][kb * 32 + grp * 16];
            const v8h ahi = *(const v8h*)&hbuf[cur][laneLo][kb * 32 + grp * 16 + 8];
            const v16h a = __builtin_shufflevector(alo, ahi,
                0, 1, 2, 3, 4, 5, 6, 7, 8, 9, 10, 11, 12, 13, 14, 15);
            acc0 = __builtin_amdgcn_wmma_f32_16x16x32_f16(
                       false, a, false, Breg[0][kb], (short)0, acc0, false, false);
            acc1 = __builtin_amdgcn_wmma_f32_16x16x32_f16(
                       false, a, false, Breg[1][kb], (short)0, acc1, false, false);
        }

        // epilogue: + x_t * W_ih^T + bias, ReLU, f16 store to next buffer
        float xv[8];
#pragma unroll
        for (int r = 0; r < 8; ++r)
            xv[r] = x[(bBase + rbase + r) * TSTEPS + t];

        const int jt0 = j0 + laneLo;
        const int jt1 = j0 + 16 + laneLo;
#pragma unroll
        for (int r = 0; r < 8; ++r) {
            float p0 = acc0[r] + xv[r] * wj[0] + bj[0];
            float p1 = acc1[r] + xv[r] * wj[1] + bj[1];
            p0 = fmaxf(p0, 0.0f);
            p1 = fmaxf(p1, 0.0f);
            hbuf[nxt][rbase + r][jt0] = (_Float16)p0;
            hbuf[nxt][rbase + r][jt1] = (_Float16)p1;
            if (t == TSTEPS - 1) {      // keep f32 copy of h_T for classifier
                hf[rbase + r][jt0] = p0;
                hf[rbase + r][jt1] = p1;
            }
        }
        __syncthreads();                // one block barrier per step
    }

    // ---- classifier: logits = h_T @ W_out^T + b_out ----
    if (tid < BM * 10) {
        const int row = tid / 10;
        const int col = tid % 10;
        float acc = b_out[col];
        const float* wr = W_out + col * HDIM;
#pragma unroll 4
        for (int k = 0; k < HDIM; ++k)
            acc += hf[row][k] * wr[k];
        logitsS[row][col] = acc;
    }
    __syncthreads();

    // ---- log_softmax per batch row ----
    if (tid < BM) {
        float m = logitsS[tid][0];
#pragma unroll
        for (int c = 1; c < 10; ++c) m = fmaxf(m, logitsS[tid][c]);
        float s = 0.0f;
#pragma unroll
        for (int c = 0; c < 10; ++c) s += __expf(logitsS[tid][c] - m);
        const float lse = m + __logf(s);
#pragma unroll
        for (int c = 0; c < 10; ++c)
            out[(bBase + tid) * 10 + c] = logitsS[tid][c] - lse;
    }
}

extern "C" void kernel_launch(void* const* d_in, const int* in_sizes, int n_in,
                              void* d_out, int out_size, void* d_ws, size_t ws_size,
                              hipStream_t stream) {
    const float* x     = (const float*)d_in[0];
    const float* W_ih  = (const float*)d_in[1];
    const float* b_ih  = (const float*)d_in[2];
    const float* W_hh  = (const float*)d_in[3];
    const float* b_hh  = (const float*)d_in[4];
    const float* W_out = (const float*)d_in[5];
    const float* b_out = (const float*)d_in[6];
    float* out = (float*)d_out;

    irnn_persistent_kernel<<<NBLOCKS, NTHREADS, 0, stream>>>(
        x, W_ih, b_ih, W_hh, b_hh, W_out, b_out, out);
}
